// MultiHeadGraphAtten_64364379898183
// MI455X (gfx1250) — matrix operand order repaired and checked by reading, not compiled, not run on hardware
//
#include <hip/hip_runtime.h>
#include <hip/hip_bf16.h>

// ---------------------------------------------------------------------------
// MultiHeadGraphAtten for MI455X (gfx1250, wave32, WMMA)
//   B=4, N=2048, IN_F=512, D=64, H=8
//   V = X@Wv + bv ; q,k = per-node/head scalars ; A = softmax_j(LR(q_j+k_i))
//   out[b,i,h*64+f] = sum_j A[b,h,i,j] * V[b,j,f]
// Dominant GEMM (A@V, 17 GFLOP) runs on v_wmma_f32_16x16x32_f16 with the
// attention weights synthesized directly in A-fragment register layout.
// Each wave owns a unique 16-row subtile and issues 4 WMMAs per A-fragment
// (one per 16-column feature subtile) so the exp stream is amortized 4x.
// ---------------------------------------------------------------------------

typedef __attribute__((ext_vector_type(16))) _Float16 v16h;
typedef __attribute__((ext_vector_type(8)))  float    v8f;

#define BB   4
#define NN   2048
#define INF  512
#define DD   64
#define HH   8
#define NSLOPE 0.01f

union Frag { v16h h; unsigned int u[8]; };

__device__ __forceinline__ float lrelu(float x) { return x > 0.0f ? x : NSLOPE * x; }

// ---------------------------------------------------------------------------
// Stage 1: V[b,n,d] = X[b,n,:] @ Wv[:,d] + bv[d]   (f32)
//          also writes Vt_h[b][d][n] = (f16)V  (transposed, packed for B-frags)
// ---------------------------------------------------------------------------
__global__ void __launch_bounds__(256) v_proj_kernel(
    const float* __restrict__ X, const float* __restrict__ Wv,
    const float* __restrict__ bv,
    float* __restrict__ V, _Float16* __restrict__ Vth)
{
    const int tid = threadIdx.x;
    const int d   = tid & 63;
    const int row = blockIdx.x * 4 + (tid >> 6);       // 0 .. B*N-1
    const float* xr = X + (size_t)row * INF;
    float acc = bv[d];
#pragma unroll 8
    for (int c = 0; c < INF; ++c)
        acc = fmaf(xr[c], Wv[c * DD + d], acc);
    V[(size_t)row * DD + d] = acc;
    const int b = row >> 11, n = row & (NN - 1);
    Vth[((size_t)(b * DD + d)) * NN + n] = (_Float16)acc;
}

// ---------------------------------------------------------------------------
// Stage 2: q[b,h,n] = V[b,n,:] @ Wq[:,h] + bq[h] ; same for k. W staged in LDS.
// ---------------------------------------------------------------------------
__global__ void __launch_bounds__(256) qk_proj_kernel(
    const float* __restrict__ V,
    const float* __restrict__ Wq, const float* __restrict__ bq,
    const float* __restrict__ Wk, const float* __restrict__ bk,
    float* __restrict__ Q, float* __restrict__ Kv)
{
    __shared__ float wq_s[DD * HH];
    __shared__ float wk_s[DD * HH];
    const int tid = threadIdx.x;
    for (int i = tid; i < DD * HH; i += 256) { wq_s[i] = Wq[i]; wk_s[i] = Wk[i]; }
    __syncthreads();

    const int row = blockIdx.x * 256 + tid;            // 0 .. B*N-1
    float qa[HH], ka[HH];
#pragma unroll
    for (int h = 0; h < HH; ++h) { qa[h] = bq[h]; ka[h] = bk[h]; }
#pragma unroll 4
    for (int d = 0; d < DD; ++d) {
        const float v = V[(size_t)row * DD + d];
#pragma unroll
        for (int h = 0; h < HH; ++h) {
            qa[h] = fmaf(v, wq_s[d * HH + h], qa[h]);
            ka[h] = fmaf(v, wk_s[d * HH + h], ka[h]);
        }
    }
    const int b = row >> 11, n = row & (NN - 1);
#pragma unroll
    for (int h = 0; h < HH; ++h) {
        Q [((size_t)(b * HH + h)) * NN + n] = qa[h];
        Kv[((size_t)(b * HH + h)) * NN + n] = ka[h];
    }
}

// ---------------------------------------------------------------------------
// Stage 3: fused attention. grid = B*H*(N/128); block = 256 (8 waves).
// Block owns rows [i0, i0+128) x all 64 features of one (b,h).
// Wave w owns row-subtile w (16 rows) and ALL 4 feature subtiles:
// per K=32 j-step it synthesizes ONE A-fragment
//   w_ij = exp(LR(q_j + k_i) - LR(qmax + k_i))   (exact softmax shift: LR is
//   monotone, so the row max is LR(qmax + k_i) -- no online rescale needed),
// accumulates the denominator alongside, loads 4 B-fragments from the
// transposed f16 V, and issues 4 independent v_wmma_f32_16x16x32_f16.
// ---------------------------------------------------------------------------
__global__ void __launch_bounds__(256) attn_kernel(
    const float* __restrict__ Q, const float* __restrict__ Kv,
    const unsigned int* __restrict__ Vth_u32,   // (b*64+f)*1024 + j/2
    float* __restrict__ out)
{
    __shared__ float q_s[NN];
    __shared__ float k_s[128];
    __shared__ float red_s[256];
    __shared__ float d_s[256];                  // 8 row-subtiles x 32

    const int tid  = threadIdx.x;
    const int tile = blockIdx.x & 15;           // i-tile (N/128 = 16)
    const int bh   = blockIdx.x >> 4;
    const int b    = bh >> 3;
    const int h    = bh & 7;
    const int i0   = tile * 128;

    const size_t qoff = (size_t)bh * NN;

    // stage q vector to LDS + block-reduce qmax
    float lm = -3.0e38f;
    for (int j = tid; j < NN; j += 256) {
        const float v = Q[qoff + j];
        q_s[j] = v;
        lm = fmaxf(lm, v);
    }
    if (tid < 128) k_s[tid] = Kv[qoff + i0 + tid];
    red_s[tid] = lm;
    __syncthreads();
#pragma unroll
    for (int s = 128; s > 0; s >>= 1) {
        if (tid < s) red_s[tid] = fmaxf(red_s[tid], red_s[tid + s]);
        __syncthreads();
    }
    const float qmax = red_s[0];

    const int wave = tid >> 5;                  // row subtile 0..7
    const int lane = tid & 31;
    const int grp  = lane >> 4;                 // A-frag K-half selector
    const int am   = lane & 15;                 // A-frag row within subtile
    const int bn   = lane & 15;                 // B-frag column (feature)
    const int bg   = lane >> 4;                 // B-frag K-half selector

    const float k_i = k_s[wave * 16 + am];
    const float m_i = lrelu(qmax + k_i);        // exact row max of LR(q_j+k_i)

    // dword bases of the 4 B-fragment columns this lane supplies
    size_t vt[4];
#pragma unroll
    for (int s = 0; s < 4; ++s)
        vt[s] = (size_t)(b * DD + 16 * s + bn) * (NN / 2) + bg * 8;

    v8f acc0 = {}, acc1 = {}, acc2 = {}, acc3 = {};
    float dpart = 0.0f;

    for (int j0 = 0; j0 < NN; j0 += 32) {
        Frag af, bf0, bf1, bf2, bf3;
        // ---- A fragment: 16x32 f16 weights, k = 16*grp + element (contiguous) ----
#pragma unroll
        for (int v = 0; v < 8; ++v) {
            const int kk = 2 * v + grp * 16;
            const int j  = j0 + kk;
            const float w0 = __expf(lrelu(q_s[j]     + k_i) - m_i);
            const float w1 = __expf(lrelu(q_s[j + 1] + k_i) - m_i);
            af.h[2 * v]     = (_Float16)w0;
            af.h[2 * v + 1] = (_Float16)w1;
            dpart += w0 + w1;
        }
        // ---- B fragments: 32x16 f16 from transposed V (pairs contiguous) ----
        const size_t jh = (size_t)(j0 >> 1);
#pragma unroll
        for (int v = 0; v < 8; ++v) {
            bf0.u[v] = Vth_u32[vt[0] + jh + v];
            bf1.u[v] = Vth_u32[vt[1] + jh + v];
            bf2.u[v] = Vth_u32[vt[2] + jh + v];
            bf3.u[v] = Vth_u32[vt[3] + jh + v];
        }
        acc0 = __builtin_amdgcn_wmma_f32_16x16x32_f16(
            false, af.h, false, bf0.h, (short)0, acc0, false, false);
        acc1 = __builtin_amdgcn_wmma_f32_16x16x32_f16(
            false, af.h, false, bf1.h, (short)0, acc1, false, false);
        acc2 = __builtin_amdgcn_wmma_f32_16x16x32_f16(
            false, af.h, false, bf2.h, (short)0, acc2, false, false);
        acc3 = __builtin_amdgcn_wmma_f32_16x16x32_f16(
            false, af.h, false, bf3.h, (short)0, acc3, false, false);
    }

    // denominator: lane m holds k 0-15 (+32k), lane m+16 holds k 16-31
    d_s[wave * 32 + lane] = dpart;
    __syncthreads();

    // ---- store: C/D layout — VGPR reg: M = reg + grp*8, N = lane&15 ----
#pragma unroll
    for (int reg = 0; reg < 8; ++reg) {
        const int mrow  = reg + grp * 8;                  // row in 16-subtile
        const int i_loc = wave * 16 + mrow;
        const float den = d_s[wave * 32 + mrow] + d_s[wave * 32 + 16 + mrow];
        const float inv = 1.0f / den;
        const int fc = lane & 15;
        const size_t base = ((size_t)(b * NN + i0 + i_loc)) * (HH * DD) + h * DD;
        out[base + fc]      = acc0[reg] * inv;
        out[base + 16 + fc] = acc1[reg] * inv;
        out[base + 32 + fc] = acc2[reg] * inv;
        out[base + 48 + fc] = acc3[reg] * inv;
    }
}

// ---------------------------------------------------------------------------
extern "C" void kernel_launch(void* const* d_in, const int* in_sizes, int n_in,
                              void* d_out, int out_size, void* d_ws, size_t ws_size,
                              hipStream_t stream) {
    const float* X  = (const float*)d_in[0];
    const float* Wv = (const float*)d_in[1];
    const float* bv = (const float*)d_in[2];
    const float* Wq = (const float*)d_in[3];
    const float* bq = (const float*)d_in[4];
    const float* Wk = (const float*)d_in[5];
    const float* bk = (const float*)d_in[6];
    float* out = (float*)d_out;

    // workspace layout
    char* ws = (char*)d_ws;
    float*     V   = (float*)ws;                                   // 2 MiB
    _Float16*  Vth = (_Float16*)(ws + (size_t)BB * NN * DD * 4);   // 1 MiB
    float*     Qs  = (float*)(ws + (size_t)BB * NN * DD * 4
                                 + (size_t)BB * NN * DD * 2);      // 256 KiB
    float*     Ks  = Qs + (size_t)BB * HH * NN;                    // 256 KiB

    v_proj_kernel<<<(BB * NN) / 4, 256, 0, stream>>>(X, Wv, bv, V, Vth);
    qk_proj_kernel<<<(BB * NN) / 256, 256, 0, stream>>>(V, Wq, bq, Wk, bk, Qs, Ks);
    attn_kernel<<<BB * HH * (NN / 128), 256, 0, stream>>>(
        Qs, Ks, (const unsigned int*)Vth, out);
}
